// NeuralVirtualizationLayer_71768903516454
// MI455X (gfx1250) — matrix operand order repaired and checked by reading, not compiled, not run on hardware
//
#include <hip/hip_runtime.h>
#include <stdint.h>
#include <math.h>

// ---------------------------------------------------------------------------
// Types for CDNA5 WMMA (gfx1250, wave32): v_wmma_f32_16x16x32_bf16
// ---------------------------------------------------------------------------
typedef __bf16 bf16_t;
typedef __attribute__((ext_vector_type(16))) __bf16 v16bf;
typedef __attribute__((ext_vector_type(8)))  __bf16 v8bf;
typedef __attribute__((ext_vector_type(8)))  float  v8f;
typedef __attribute__((ext_vector_type(8)))  float  v8ff;

#define B_    64
#define IN_   1024
#define HID_  2048
#define AMAX  256
#define MEMN  4096
#define RR    512
#define DD    512
#define G3R   1536   // 3*R

__device__ __forceinline__ int lane_id() { return (int)(threadIdx.x & 31); }

__device__ __forceinline__ v16bf cat8(v8bf lo, v8bf hi) {
  return __builtin_shufflevector(lo, hi, 0, 1, 2, 3, 4, 5, 6, 7, 8, 9, 10, 11,
                                 12, 13, 14, 15);
}

// ---------------------------------------------------------------------------
// Fragment builders (contiguous-K order, same for A and B). Per-lane data is contiguous in
// K, so fragments are built from aligned 16B vector loads (b128) with
// f32->bf16 via convertvector where the source is f32.
// A (16x32): lanes 0-15 row M=lane, K 0..15; lanes 16-31 row M, K 16..31
// B (32x16): lanes 0-15 col N=lane, K 0..15; lanes 16-31 col N, K 16..31
// C/D: VGPR r, lane<16 -> (M=r, N=lane); lane>=16 -> (M=r+8, N=lane-16)
// ---------------------------------------------------------------------------
__device__ __forceinline__ v16bf frag_a(const bf16_t* A, long lda, long row0,
                                        long k0) {
  int lane = lane_id();
  const bf16_t* p =
      A + (row0 + (lane & 15)) * lda + k0 + ((lane >> 4) ? 16 : 0);
  v8bf lo = __builtin_bit_cast(v8bf, *(const uint4*)p);
  v8bf hi = __builtin_bit_cast(v8bf, *(const uint4*)(p + 8));
  return cat8(lo, hi);
}

__device__ __forceinline__ v16bf frag_a(const float* A, long lda, long row0,
                                        long k0) {
  int lane = lane_id();
  const float* p = A + (row0 + (lane & 15)) * lda + k0 + ((lane >> 4) ? 16 : 0);
  v8ff lo = *(const v8ff*)p;
  v8ff hi = *(const v8ff*)(p + 8);
  return cat8(__builtin_convertvector(lo, v8bf),
              __builtin_convertvector(hi, v8bf));
}

// B operand from weight stored (N x K) row-major: B[k][n] = W[n][k]
__device__ __forceinline__ v16bf frag_b_nk(const bf16_t* W, long ldw, long n0,
                                           long k0) {
  int lane = lane_id();
  const bf16_t* p =
      W + (n0 + (lane & 15)) * ldw + k0 + ((lane >> 4) ? 16 : 0);
  v8bf lo = __builtin_bit_cast(v8bf, *(const uint4*)p);
  v8bf hi = __builtin_bit_cast(v8bf, *(const uint4*)(p + 8));
  return cat8(lo, hi);
}

__device__ __forceinline__ v16bf frag_b_nk(const float* W, long ldw, long n0,
                                           long k0) {
  int lane = lane_id();
  const float* p = W + (n0 + (lane & 15)) * ldw + k0 + ((lane >> 4) ? 16 : 0);
  v8ff lo = *(const v8ff*)p;
  v8ff hi = *(const v8ff*)(p + 8);
  return cat8(__builtin_convertvector(lo, v8bf),
              __builtin_convertvector(hi, v8bf));
}

// B operand from matrix stored (K x N) row-major (strided -> scalar gathers)
template<typename T>
__device__ __forceinline__ v16bf frag_b_kn(const T* Bm, long ldb, long n0,
                                           long k0) {
  int lane = lane_id();
  long n = n0 + (lane & 15);
  int kb = (lane >> 4) ? 16 : 0;
  v16bf f;
#pragma unroll
  for (int j = 0; j < 16; ++j)
    f[j] = (bf16_t)(float)Bm[(k0 + kb + j) * ldb + n];
  return f;
}

__device__ __forceinline__ v8f wmma_bf(v16bf a, v16bf b, v8f c) {
  return __builtin_amdgcn_wmma_f32_16x16x32_bf16(false, a, false, b, (short)0, c,
                                                 false, false);
}

__device__ __forceinline__ float sigmoidf_(float x) {
  return 1.0f / (1.0f + __expf(-x));
}

// ---------------------------------------------------------------------------
// K0: prep — bf16 copies of x and active hidden rows; full hidden -> out.
// ---------------------------------------------------------------------------
__global__ void prep_kernel(const float* __restrict__ x,
                            const float* __restrict__ hidden,
                            const int* __restrict__ anp,
                            bf16_t* __restrict__ x_bf,
                            bf16_t* __restrict__ curr_bf,
                            float* __restrict__ out_hidden) {
  int A = *anp;
  long i = (long)blockIdx.x * blockDim.x + threadIdx.x;
  if (i < (long)B_ * IN_) x_bf[i] = (bf16_t)x[i];
  long total = (long)B_ * AMAX * RR;
  if (i >= total) return;
  float h = hidden[i];
  out_hidden[i] = h;
  long a = (i / RR) % AMAX;
  curr_bf[i] = (a < A) ? (bf16_t)h : (bf16_t)0.0f;
}

// ---------------------------------------------------------------------------
// K1: mmu GEMM — curr(16384x512) @ mmu_w^T, only cols [0,1537): q|k|v|gate
// ---------------------------------------------------------------------------
__global__ void mmu_kernel(const bf16_t* __restrict__ curr,
                           const float* __restrict__ mmu_w,
                           const float* __restrict__ mmu_b,
                           bf16_t* __restrict__ qb, bf16_t* __restrict__ kb_,
                           bf16_t* __restrict__ vb, float* __restrict__ wgate) {
  const int NT = 97;                       // 97*16 = 1552 >= 1537 cols
  const int MT = (B_ * AMAX) >> 4;
  int wave = blockIdx.x * 8 + (int)(threadIdx.x >> 5);
  if (wave >= MT * NT) return;
  long mt = wave / NT, nt = wave % NT;
  long row0 = mt << 4, n0 = nt << 4;
  v8f acc = {};
  for (int k = 0; k < RR; k += 32)
    acc = wmma_bf(frag_a(curr, (long)RR, row0, k),
                  frag_b_nk(mmu_w, (long)RR, n0, k), acc);
  int lane = lane_id();
  long n = n0 + (lane & 15);
  int rb = (lane >> 4) ? 8 : 0;
  float bv = mmu_b[n];
#pragma unroll
  for (int r = 0; r < 8; ++r) {
    long row = row0 + r + rb;
    float v = acc[r] + bv;
    if (n < DD)               qb[row * DD + n] = (bf16_t)v;
    else if (n < 2 * DD)      kb_[row * DD + (n - DD)] = (bf16_t)v;
    else if (n < 3 * DD)      vb[row * DD + (n - 2 * DD)] = (bf16_t)v;
    else if (n == 3 * DD)     wgate[row] = sigmoidf_(v);
  }
}

// ---------------------------------------------------------------------------
// K2/K4: softmax stats (running max & sum over M) for scores = Q @ mem^T
// grid (B_, AMAX/32), block 256
// ---------------------------------------------------------------------------
__global__ void attn_stats_kernel(const bf16_t* __restrict__ Q,
                                  const float* __restrict__ memory,
                                  float* __restrict__ stats, float scale) {
  int b = blockIdx.x;
  long a0 = (long)blockIdx.y * 32;
  const bf16_t* Qb = Q + (long)b * AMAX * DD;
  const float* Mb = memory + (long)b * MEMN * DD;
  __shared__ float s[32][64];
  int wave = (int)(threadIdx.x >> 5);
  int rt = wave & 1, ct = wave >> 1;
  int lane = lane_id();
  float runm = -1e30f, runl = 0.0f;
  for (int mc = 0; mc < MEMN; mc += 64) {
    if (threadIdx.x < 64 && mc + 64 < MEMN)
      __builtin_prefetch(Mb + (long)(mc + 64 + (int)threadIdx.x) * DD, 0, 0);
    v8f acc = {};
    for (int k = 0; k < DD; k += 32)
      acc = wmma_bf(frag_a(Qb, (long)DD, a0 + 16 * rt, k),
                    frag_b_nk(Mb, (long)DD, (long)mc + 16 * ct, k), acc);
    int col = lane & 15, rb = (lane >> 4) ? 8 : 0;
#pragma unroll
    for (int r = 0; r < 8; ++r)
      s[16 * rt + r + rb][16 * ct + col] = acc[r] * scale;
    __syncthreads();
    if (threadIdx.x < 32) {
      int rr = (int)threadIdx.x;
      float cm = -1e30f;
      for (int c = 0; c < 64; ++c) cm = fmaxf(cm, s[rr][c]);
      float nm = fmaxf(runm, cm);
      float add = 0.0f;
      for (int c = 0; c < 64; ++c) add += __expf(s[rr][c] - nm);
      runl = runl * __expf(runm - nm) + add;
      runm = nm;
    }
    __syncthreads();
  }
  if (threadIdx.x < 32) {
    long a = a0 + threadIdx.x;
    stats[((long)b * AMAX + a) * 2 + 0] = runm;
    stats[((long)b * AMAX + a) * 2 + 1] = runl;
  }
}

// ---------------------------------------------------------------------------
// K3: read attention pass 2 — read = softmax(Q memT) @ mem, stats precomputed.
// grid (B_, AMAX/32), block 256; 32 a-rows per block, full D=512 accumulators.
// ---------------------------------------------------------------------------
__global__ void attn_read_kernel(const bf16_t* __restrict__ Q,
                                 const float* __restrict__ memory,
                                 const float* __restrict__ stats,
                                 bf16_t* __restrict__ outbf, float scale) {
  int b = blockIdx.x;
  long a0 = (long)blockIdx.y * 32;
  const bf16_t* Qb = Q + (long)b * AMAX * DD;
  const float* Mb = memory + (long)b * MEMN * DD;
  const float* st = stats + ((long)b * AMAX + a0) * 2;
  __shared__ float  s[32][32];
  __shared__ bf16_t p[32][32];
  __shared__ float  lmax[32], linv[32];
  if (threadIdx.x < 32) {
    lmax[threadIdx.x] = st[threadIdx.x * 2];
    linv[threadIdx.x] = 1.0f / st[threadIdx.x * 2 + 1];
  }
  __syncthreads();
  int wave = (int)(threadIdx.x >> 5);
  int lane = lane_id();
  int rt = wave & 1;
  int ctb = (wave >> 1) * 8;                 // 8 d-tiles per wave -> 512 cols
  v8f acc[8];
#pragma unroll
  for (int t = 0; t < 8; ++t) { v8f z = {}; acc[t] = z; }
  for (int mc = 0; mc < MEMN; mc += 32) {
    if (threadIdx.x < 32 && mc + 32 < MEMN)
      __builtin_prefetch(Mb + (long)(mc + 32 + (int)threadIdx.x) * DD, 0, 0);
    if (wave < 4) {
      int srt = wave & 1, sct = wave >> 1;
      v8f sa = {};
      for (int k = 0; k < DD; k += 32)
        sa = wmma_bf(frag_a(Qb, (long)DD, a0 + 16 * srt, k),
                     frag_b_nk(Mb, (long)DD, (long)mc + 16 * sct, k), sa);
      int col = lane & 15, rb = (lane >> 4) ? 8 : 0;
#pragma unroll
      for (int r = 0; r < 8; ++r)
        s[16 * srt + r + rb][16 * sct + col] = sa[r] * scale;
    }
    __syncthreads();
    for (int e = (int)threadIdx.x; e < 32 * 32; e += 256) {
      int rr = e >> 5, cc = e & 31;
      p[rr][cc] = (bf16_t)(__expf(s[rr][cc] - lmax[rr]) * linv[rr]);
    }
    __syncthreads();
#pragma unroll
    for (int t = 0; t < 8; ++t)
      acc[t] = wmma_bf(frag_a(&p[0][0], 32L, (long)(16 * rt), 0L),
                       frag_b_kn(Mb, (long)DD, (long)(16 * (ctb + t)), (long)mc),
                       acc[t]);
    __syncthreads();
  }
  int col = lane & 15, rb = (lane >> 4) ? 8 : 0;
#pragma unroll
  for (int t = 0; t < 8; ++t) {
    long n0 = 16 * (ctb + t);
#pragma unroll
    for (int r = 0; r < 8; ++r)
      outbf[((long)b * AMAX + a0 + 16 * rt + r + rb) * DD + n0 + col] =
          (bf16_t)acc[t][r];
  }
}

// ---------------------------------------------------------------------------
// Generic GEMM: C(MxN) = A_bf16(MxK,row) @ W^T (W: NxK f32 row-major) + bias
// Each wave owns 1 m-tile x 4 n-tiles: A fragment reused for 4 WMMAs.
// ---------------------------------------------------------------------------
template<bool OUT_BF>
__global__ void gemm_awt(const bf16_t* __restrict__ A, const float* __restrict__ W,
                         const float* __restrict__ bias, float* __restrict__ Cf,
                         bf16_t* __restrict__ Cb, int M, int N, int K) {
  int wave = blockIdx.x * 8 + (int)(threadIdx.x >> 5);
  int ngrp = N >> 6;                       // groups of 4 n-tiles
  if (wave >= (M >> 4) * ngrp) return;
  long mt = wave / ngrp, ng = wave % ngrp;
  long row0 = mt << 4, n0 = ng << 6;
  v8f acc[4];
#pragma unroll
  for (int t = 0; t < 4; ++t) { v8f z = {}; acc[t] = z; }
  for (int k = 0; k < K; k += 32) {
    v16bf a = frag_a(A, (long)K, row0, k);
#pragma unroll
    for (int t = 0; t < 4; ++t)
      acc[t] = wmma_bf(a, frag_b_nk(W, (long)K, n0 + 16 * t, k), acc[t]);
  }
  int lane = lane_id();
  int col = lane & 15, rb = (lane >> 4) ? 8 : 0;
#pragma unroll
  for (int t = 0; t < 4; ++t) {
    long nn = n0 + 16 * t + col;
    float bv = bias ? bias[nn] : 0.0f;
#pragma unroll
    for (int r = 0; r < 8; ++r) {
      long row = row0 + r + rb;
      float v = acc[t][r] + bv;
      if constexpr (OUT_BF) Cb[row * N + nn] = (bf16_t)v;
      else                  Cf[row * N + nn] = v;
    }
  }
}

// ---------------------------------------------------------------------------
// K6: concat [x (broadcast over a) | read_data] -> gru_in (bf16, width 1536)
// ---------------------------------------------------------------------------
__global__ void concat_kernel(const bf16_t* __restrict__ x_bf,
                              const bf16_t* __restrict__ read_bf,
                              bf16_t* __restrict__ grin) {
  long i = (long)blockIdx.x * blockDim.x + threadIdx.x;
  if (i >= (long)B_ * AMAX * G3R) return;
  long j = i % G3R, ba = i / G3R, b = ba / AMAX;
  grin[i] = (j < IN_) ? x_bf[b * IN_ + j] : read_bf[ba * DD + (j - IN_)];
}

// ---------------------------------------------------------------------------
// K8: GRU elementwise; writes new hidden (f32 out) + bf16 copy for next GEMMs
// ---------------------------------------------------------------------------
__global__ void gru_kernel(const float* __restrict__ gi, const float* __restrict__ gh,
                           const float* __restrict__ hidden, const int* __restrict__ anp,
                           float* __restrict__ out_hidden, bf16_t* __restrict__ new_bf) {
  int A = *anp;
  long i = (long)blockIdx.x * blockDim.x + threadIdx.x;
  if (i >= (long)B_ * AMAX * RR) return;
  long r = i % RR, ba = i / RR, a = ba % AMAX;
  float h = hidden[i], nv;
  if (a < A) {
    long base = ba * G3R;
    float r_ = sigmoidf_(gi[base + r] + gh[base + r]);
    float z_ = sigmoidf_(gi[base + RR + r] + gh[base + RR + r]);
    float n_ = tanhf(gi[base + 2 * RR + r] + r_ * gh[base + 2 * RR + r]);
    nv = (1.0f - z_) * n_ + z_ * h;
    out_hidden[i] = nv;
  } else {
    nv = h;                                  // rows >= A keep old hidden
  }
  new_bf[i] = (bf16_t)nv;
}

// ---------------------------------------------------------------------------
// K12: memory update. grid (B_, MEMN/32). sT = mem-tile @ k^T -> gated masked
// write-attn p (bf16 LDS) -> total_weight row-sum -> combined = p @ v -> blend.
// ---------------------------------------------------------------------------
__global__ void mem_update_kernel(const bf16_t* __restrict__ Kmat,
                                  const bf16_t* __restrict__ Vmat,
                                  const float* __restrict__ wgate,
                                  const float* __restrict__ stats,
                                  const float* __restrict__ memory,
                                  const int* __restrict__ anp,
                                  float* __restrict__ newmem, float scale) {
  int b = blockIdx.x, A = *anp;
  long m0 = (long)blockIdx.y * 32;
  const bf16_t* Kb = Kmat + (long)b * AMAX * DD;
  const bf16_t* Vb = Vmat + (long)b * AMAX * DD;
  const float* Mb = memory + (long)b * MEMN * DD;
  float* Ob = newmem + (long)b * MEMN * DD;
  const float* st = stats + (long)b * AMAX * 2;
  const float* wg = wgate + (long)b * AMAX;
  __shared__ bf16_t pT[32][AMAX];
  __shared__ float  tw[32];
  int wave = (int)(threadIdx.x >> 5);
  int lane = lane_id();
  // phase 1: sT (32 x 256) — 32 tiles, 4 per wave
  {
    int rt = wave & 1, ctb = (wave >> 1) * 4;
    int col = lane & 15, rb = (lane >> 4) ? 8 : 0;
#pragma unroll
    for (int t = 0; t < 4; ++t) {
      v8f acc = {};
      for (int k = 0; k < DD; k += 32)
        acc = wmma_bf(frag_a(Mb, (long)DD, m0 + 16 * rt, k),
                      frag_b_nk(Kb, (long)DD, (long)(16 * (ctb + t)), k), acc);
      int a = 16 * (ctb + t) + col;
      float mx = st[a * 2], linv = 1.0f / st[a * 2 + 1], g = wg[a];
#pragma unroll
      for (int r = 0; r < 8; ++r) {
        float pv = (a < A) ? g * __expf(acc[r] * scale - mx) * linv : 0.0f;
        pT[16 * rt + r + rb][a] = (bf16_t)pv;
      }
    }
  }
  __syncthreads();
  if (threadIdx.x < 32) {
    float sum = 0.0f;
    for (int c = 0; c < AMAX; ++c) sum += (float)pT[threadIdx.x][c];
    tw[threadIdx.x] = fminf(fmaxf(sum, 0.0f), 1.0f);
  }
  __syncthreads();
  // phase 2: combined (32 x 512) = pT @ V — 64 tiles, 8 per wave
  {
    int rt = wave & 1, ctb = (wave >> 1) * 8;
    v8f acc[8];
#pragma unroll
    for (int t = 0; t < 8; ++t) { v8f z = {}; acc[t] = z; }
    for (int k = 0; k < AMAX; k += 32) {
      v16bf a = frag_a(&pT[0][0], (long)AMAX, (long)(16 * rt), (long)k);
#pragma unroll
      for (int t = 0; t < 8; ++t)
        acc[t] = wmma_bf(a, frag_b_kn(Vb, (long)DD, (long)(16 * (ctb + t)),
                                      (long)k),
                         acc[t]);
    }
    int col = lane & 15, rb = (lane >> 4) ? 8 : 0;
#pragma unroll
    for (int t = 0; t < 8; ++t) {
      long n0 = 16 * (ctb + t);
#pragma unroll
      for (int r = 0; r < 8; ++r) {
        int rloc = 16 * rt + r + rb;
        long gidx = (m0 + rloc) * (long)DD + n0 + col;
        Ob[gidx] = Mb[gidx] * (1.0f - tw[rloc]) + acc[t][r];
      }
    }
  }
}

// ---------------------------------------------------------------------------
// K11: agg attention over NPUs + LayerNorm. One block per batch, 256 threads.
// ---------------------------------------------------------------------------
__global__ void agg_ln_kernel(const float* __restrict__ proj,
                              const float* __restrict__ aggq,
                              const int* __restrict__ anp,
                              const float* __restrict__ gamma,
                              const float* __restrict__ beta,
                              float* __restrict__ out) {
  int b = blockIdx.x, A = *anp, tid = (int)threadIdx.x;
  const float* P = proj + (long)b * AMAX * HID_;
  __shared__ float sc[AMAX];
  __shared__ float red[256];
  __shared__ float aggv[HID_];
  float s = -1e30f;
  if (tid < A) {
    float acc = 0.0f;
    for (int h = 0; h < HID_; ++h) acc += aggq[h] * P[(long)tid * HID_ + h];
    s = acc * (1.0f / sqrtf((float)HID_));
  }
  sc[tid] = s;
  red[tid] = s;
  __syncthreads();
  for (int o = 128; o > 0; o >>= 1) {
    if (tid < o) red[tid] = fmaxf(red[tid], red[tid + o]);
    __syncthreads();
  }
  float mx = red[0];
  __syncthreads();
  float e = (tid < A) ? __expf(sc[tid] - mx) : 0.0f;
  red[tid] = e;
  __syncthreads();
  for (int o = 128; o > 0; o >>= 1) {
    if (tid < o) red[tid] += red[tid + o];
    __syncthreads();
  }
  float denom = red[0];
  __syncthreads();
  sc[tid] = e / denom;
  __syncthreads();
  for (int h = tid; h < HID_; h += 256) {
    float acc = 0.0f;
    for (int a = 0; a < A; ++a) acc += sc[a] * P[(long)a * HID_ + h];
    aggv[h] = acc;
  }
  __syncthreads();
  float ls = 0.0f, ls2 = 0.0f;
  for (int h = tid; h < HID_; h += 256) { float v = aggv[h]; ls += v; ls2 += v * v; }
  red[tid] = ls; __syncthreads();
  for (int o = 128; o > 0; o >>= 1) { if (tid < o) red[tid] += red[tid + o]; __syncthreads(); }
  float mu = red[0] / HID_;
  __syncthreads();
  red[tid] = ls2; __syncthreads();
  for (int o = 128; o > 0; o >>= 1) { if (tid < o) red[tid] += red[tid + o]; __syncthreads(); }
  float var = red[0] / HID_ - mu * mu;
  float inv = rsqrtf(var + 1e-5f);
  for (int h = tid; h < HID_; h += 256)
    out[(long)b * HID_ + h] = (aggv[h] - mu) * inv * gamma[h] + beta[h];
}

// ---------------------------------------------------------------------------
// Host launcher
// ---------------------------------------------------------------------------
extern "C" void kernel_launch(void* const* d_in, const int* in_sizes, int n_in,
                              void* d_out, int out_size, void* d_ws, size_t ws_size,
                              hipStream_t stream) {
  (void)in_sizes; (void)n_in; (void)out_size; (void)ws_size;
  const float* x        = (const float*)d_in[0];
  const float* hidden   = (const float*)d_in[1];
  const float* memory   = (const float*)d_in[2];
  const float* mmu_w    = (const float*)d_in[3];
  const float* mmu_b    = (const float*)d_in[4];
  const float* gru_w_ih = (const float*)d_in[5];
  const float* gru_b_ih = (const float*)d_in[6];
  const float* gru_w_hh = (const float*)d_in[7];
  const float* gru_b_hh = (const float*)d_in[8];
  const float* out_w    = (const float*)d_in[9];
  const float* out_b    = (const float*)d_in[10];
  const float* agg_q    = (const float*)d_in[11];
  const float* proj_w   = (const float*)d_in[12];
  const float* proj_b   = (const float*)d_in[13];
  const float* ln_gamma = (const float*)d_in[14];
  const float* ln_beta  = (const float*)d_in[15];
  const int*   anp      = (const int*)d_in[16];

  float* out_y   = (float*)d_out;                       // (B, HID)
  float* out_hid = out_y + (size_t)B_ * HID_;           // (B, AMAX, R)
  float* out_mem = out_hid + (size_t)B_ * AMAX * RR;    // (B, MEM, D)

  uint8_t* w = (uint8_t*)d_ws;
  size_t off = 0;
  auto alloc = [&](size_t bytes) -> void* {
    off = (off + 255) & ~(size_t)255;
    void* p = w + off;
    off += bytes;
    return p;
  };
  const size_t BA = (size_t)B_ * AMAX;
  bf16_t* curr_bf = (bf16_t*)alloc(BA * RR * 2);
  bf16_t* x_bf    = (bf16_t*)alloc((size_t)B_ * IN_ * 2);
  bf16_t* q_bf    = (bf16_t*)alloc(BA * DD * 2);
  bf16_t* k_bf    = (bf16_t*)alloc(BA * DD * 2);
  bf16_t* v_bf    = (bf16_t*)alloc(BA * DD * 2);
  float*  wgate   = (float*)alloc(BA * 4);
  float*  rstats  = (float*)alloc(BA * 2 * 4);
  float*  wstats  = (float*)alloc(BA * 2 * 4);
  bf16_t* read_bf = (bf16_t*)alloc(BA * DD * 2);
  bf16_t* grin_bf = (bf16_t*)alloc(BA * G3R * 2);
  float*  gh      = (float*)alloc(BA * G3R * 4);
  float*  gi      = (float*)alloc(BA * G3R * 4);
  bf16_t* new_bf  = (bf16_t*)alloc(BA * RR * 2);
  bf16_t* npu_bf  = (bf16_t*)alloc(BA * IN_ * 2);
  float*  proj    = (float*)alloc(BA * HID_ * 4);

  const float inv_sqrt_d = 0.044194173824159216f;       // 1/sqrt(512)
  const int TPB = 256;

  // K0 prep
  {
    long total = (long)BA * RR;
    prep_kernel<<<dim3((unsigned)((total + TPB - 1) / TPB)), TPB, 0, stream>>>(
        x, hidden, anp, x_bf, curr_bf, out_hid);
  }
  // K1 mmu
  {
    int waves = ((B_ * AMAX) >> 4) * 97;
    mmu_kernel<<<dim3((unsigned)((waves + 7) / 8)), TPB, 0, stream>>>(
        curr_bf, mmu_w, mmu_b, q_bf, k_bf, v_bf, wgate);
  }
  // K2 read stats, K3 read pass2
  attn_stats_kernel<<<dim3(B_, AMAX / 32), TPB, 0, stream>>>(q_bf, memory, rstats,
                                                             inv_sqrt_d);
  attn_read_kernel<<<dim3(B_, AMAX / 32), TPB, 0, stream>>>(q_bf, memory, rstats,
                                                            read_bf, inv_sqrt_d);
  // K4 write stats
  attn_stats_kernel<<<dim3(B_, AMAX / 32), TPB, 0, stream>>>(k_bf, memory, wstats,
                                                             inv_sqrt_d);
  // K12 memory update
  mem_update_kernel<<<dim3(B_, MEMN / 32), TPB, 0, stream>>>(
      k_bf, v_bf, wgate, wstats, memory, anp, out_mem, inv_sqrt_d);
  // K5 gh GEMM
  {
    int waves = ((B_ * AMAX) >> 4) * (G3R >> 6);
    gemm_awt<false><<<dim3((unsigned)((waves + 7) / 8)), TPB, 0, stream>>>(
        curr_bf, gru_w_hh, gru_b_hh, gh, nullptr, B_ * AMAX, G3R, RR);
  }
  // K6 concat
  {
    long total = (long)BA * G3R;
    concat_kernel<<<dim3((unsigned)((total + TPB - 1) / TPB)), TPB, 0, stream>>>(
        x_bf, read_bf, grin_bf);
  }
  // K7 gi GEMM
  {
    int waves = ((B_ * AMAX) >> 4) * (G3R >> 6);
    gemm_awt<false><<<dim3((unsigned)((waves + 7) / 8)), TPB, 0, stream>>>(
        grin_bf, gru_w_ih, gru_b_ih, gi, nullptr, B_ * AMAX, G3R, G3R);
  }
  // K8 GRU elementwise
  {
    long total = (long)BA * RR;
    gru_kernel<<<dim3((unsigned)((total + TPB - 1) / TPB)), TPB, 0, stream>>>(
        gi, gh, hidden, anp, out_hid, new_bf);
  }
  // K9 out GEMM -> npu_bf
  {
    int waves = ((B_ * AMAX) >> 4) * (IN_ >> 6);
    gemm_awt<true><<<dim3((unsigned)((waves + 7) / 8)), TPB, 0, stream>>>(
        new_bf, out_w, out_b, nullptr, npu_bf, B_ * AMAX, IN_, RR);
  }
  // K10 proj GEMM -> proj f32
  {
    int waves = ((B_ * AMAX) >> 4) * (HID_ >> 6);
    gemm_awt<false><<<dim3((unsigned)((waves + 7) / 8)), TPB, 0, stream>>>(
        npu_bf, proj_w, proj_b, proj, nullptr, B_ * AMAX, HID_, IN_);
  }
  // K11 agg + layernorm
  agg_ln_kernel<<<dim3(B_), TPB, 0, stream>>>(proj, agg_q, anp, ln_gamma, ln_beta,
                                              out_y);
}
